// GNNNet_42520176230428
// MI455X (gfx1250) — matrix operand order repaired and checked by reading, not compiled, not run on hardware
//
#include <hip/hip_runtime.h>
#include <cstdint>
#include <cstddef>

// ---------------------------------------------------------------------------
// Problem constants (match reference)
// ---------------------------------------------------------------------------
constexpr int N    = 50000;
constexpr int E    = 400000;
constexpr int FIN  = 128;
constexpr int EDIM = 16;
constexpr int NH   = 4;      // heads
constexpr int C    = 64;     // per-head channels
constexpr int HC   = 256;    // NH*C
constexpr int NG   = 64;     // graphs
constexpr int OUTD = 10;
constexpr int EF   = E + N;  // edges incl. self loops

// ---------------------------------------------------------------------------
// WMMA types (CDNA5 gfx1250, wave32)
// ---------------------------------------------------------------------------
typedef __attribute__((ext_vector_type(16))) __bf16 v16bf;
typedef __attribute__((ext_vector_type(8)))  float  v8f;

union FragBF16 {
    v16bf v;
    uint4 q[2];
};

__device__ __forceinline__ uint16_t f2bf(float f) {
    // round-to-nearest-even fp32 -> bf16
    uint32_t u = __float_as_uint(f);
    uint32_t r = u + 0x7FFFu + ((u >> 16) & 1u);
    return (uint16_t)(r >> 16);
}

// monotone float<->uint encoding so unsigned atomicMax == float max
__device__ __forceinline__ uint32_t enc_f32(float f) {
    uint32_t u = __float_as_uint(f);
    return (u & 0x80000000u) ? ~u : (u | 0x80000000u);
}
__device__ __forceinline__ float dec_f32(uint32_t u) {
    return (u & 0x80000000u) ? __uint_as_float(u & 0x7FFFFFFFu)
                             : __uint_as_float(~u);
}

__device__ __forceinline__ float wave_reduce_add(float v) {
    #pragma unroll
    for (int off = 16; off > 0; off >>= 1) v += __shfl_xor(v, off, 32);
    return v;
}

// ---------------------------------------------------------------------------
// Utility kernels
// ---------------------------------------------------------------------------
__global__ void k_zero_u32(uint32_t* p, int n) {
    int i = blockIdx.x * blockDim.x + threadIdx.x;
    if (i < n) p[i] = 0u;
}

__global__ void k_cast_bf16(const float* __restrict__ x, uint16_t* __restrict__ o, int n) {
    int i = blockIdx.x * blockDim.x + threadIdx.x;
    if (i < n) o[i] = f2bf(x[i]);
}

// W [K, 256] fp32 -> Wt [256, K] bf16 (so B-fragment K runs contiguously)
__global__ void k_transpose_cast(const float* __restrict__ W, uint16_t* __restrict__ Wt, int K) {
    int idx = blockIdx.x * blockDim.x + threadIdx.x;
    if (idx >= K * HC) return;
    int o = idx & (HC - 1);
    int k = idx >> 8;
    Wt[(size_t)o * K + k] = f2bf(W[(size_t)k * HC + o]);
}

// ---------------------------------------------------------------------------
// Self-loop edge_attr: deg + mean of incoming edge_attr per dst
// ---------------------------------------------------------------------------
__global__ void k_deg_sum(const int* __restrict__ dst, const float* __restrict__ ea,
                          float* __restrict__ deg, float* __restrict__ sea) {
    int idx = blockIdx.x * blockDim.x + threadIdx.x;
    if (idx >= E * EDIM) return;
    int e = idx >> 4, f = idx & 15;
    int d = dst[e];
    atomicAdd(&sea[(size_t)d * EDIM + f], ea[idx]);
    if (f == 0) atomicAdd(&deg[d], 1.0f);
}

__global__ void k_mean_div(float* __restrict__ mea, const float* __restrict__ deg) {
    int idx = blockIdx.x * blockDim.x + threadIdx.x;
    if (idx >= N * EDIM) return;
    int n = idx >> 4;
    mea[idx] = mea[idx] / fmaxf(deg[n], 1.0f);
}

// ---------------------------------------------------------------------------
// WMMA GEMM: Hout[N,256] = Xb[N,K](bf16) @ W[K,256]  (W given transposed: Wt[256,K] bf16)
// Block = 256 threads = 8 waves; block covers 16 rows x 256 cols.
// Wave w: column tiles [32w,32w+16) and [32w+16,32w+32). K-loop step 32.
// Fragment layouts used here (contiguous K; same order in A and B):
//   A 16x32 bf16: lanes 0-15 M=0-15 / lanes 16-31 M=0-15; lane-half splits K 0-15 / 16-31.
//   B 32x16 bf16: lane holds one column, lane-half splits K 0-15 / 16-31.
// ---------------------------------------------------------------------------
__global__ void __launch_bounds__(256)
k_gemm_bf16(const uint16_t* __restrict__ Xb, const uint16_t* __restrict__ Wt,
            float* __restrict__ Hout, int K) {
    const int m0    = blockIdx.x * 16;
    const int wave  = threadIdx.x >> 5;
    const int lane  = threadIdx.x & 31;
    const int lrow  = lane & 15;
    const int lhalf = lane >> 4;

    const uint16_t* xrow = Xb + (size_t)(m0 + lrow) * K;
    const int c0 = wave * 32;
    const uint16_t* w0 = Wt + (size_t)(c0 + lrow) * K;
    const uint16_t* w1 = Wt + (size_t)(c0 + 16 + lrow) * K;

    v8f acc0 = {};
    v8f acc1 = {};

    for (int kb = 0; kb < K; kb += 32) {
        // A fragment: element i -> K = kb + lhalf*16 + i
        FragBF16 a;
        a.q[0] = *reinterpret_cast<const uint4*>(xrow + kb + lhalf * 16);
        a.q[1] = *reinterpret_cast<const uint4*>(xrow + kb + lhalf * 16 + 8);
        // B fragments: element i -> K = kb + lhalf*16 + i  (contiguous in Wt row)
        FragBF16 b0, b1;
        b0.q[0] = *reinterpret_cast<const uint4*>(w0 + kb + lhalf * 16);
        b0.q[1] = *reinterpret_cast<const uint4*>(w0 + kb + lhalf * 16 + 8);
        b1.q[0] = *reinterpret_cast<const uint4*>(w1 + kb + lhalf * 16);
        b1.q[1] = *reinterpret_cast<const uint4*>(w1 + kb + lhalf * 16 + 8);

        acc0 = __builtin_amdgcn_wmma_f32_16x16x32_bf16(false, a.v, false, b0.v,
                                                       (short)0, acc0, false, false);
        acc1 = __builtin_amdgcn_wmma_f32_16x16x32_bf16(false, a.v, false, b1.v,
                                                       (short)0, acc1, false, false);

        // speculative prefetch of next K-slab (global_prefetch_b8)
        __builtin_prefetch(xrow + kb + 32, 0, 1);
    }

    // C/D layout: VGPR r -> M = r + 8*lhalf ; N = c0 + (lane&15)
    float* out = Hout + (size_t)m0 * HC;
    #pragma unroll
    for (int r = 0; r < 8; ++r) {
        int rr = r + lhalf * 8;
        out[(size_t)rr * HC + c0 + lrow]      = acc0[r];
        out[(size_t)rr * HC + c0 + 16 + lrow] = acc1[r];
    }
}

// ---------------------------------------------------------------------------
// Fold We (16x256) with a_e (4x64) -> M (16x4):  M[f,h] = sum_c We[f,h*64+c]*ae[h,c]
// ---------------------------------------------------------------------------
__global__ void k_fold_M(const float* __restrict__ We, const float* __restrict__ ae,
                         float* __restrict__ M) {
    int t = threadIdx.x;
    if (t >= EDIM * NH) return;
    int f = t >> 2, h = t & 3;
    float s = 0.f;
    #pragma unroll
    for (int c = 0; c < C; ++c) s += We[(size_t)f * HC + h * C + c] * ae[h * C + c];
    M[f * NH + h] = s;
}

// per-node attention logits al_s, al_d
__global__ void k_node_logits(const float* __restrict__ h, const float* __restrict__ as_,
                              const float* __restrict__ ad_,
                              float* __restrict__ als, float* __restrict__ ald) {
    int idx = blockIdx.x * blockDim.x + threadIdx.x;
    if (idx >= N * NH) return;
    int n = idx >> 2, hh = idx & 3;
    const float* hp = h + (size_t)n * HC + hh * C;
    float s = 0.f, d = 0.f;
    #pragma unroll 8
    for (int c = 0; c < C; ++c) {
        float v = hp[c];
        s += v * as_[hh * C + c];
        d += v * ad_[hh * C + c];
    }
    als[idx] = s;
    ald[idx] = d;
}

// edge logits + leaky_relu + segment-max(dst)
__global__ void k_edge_alpha(const int* __restrict__ src, const int* __restrict__ dst,
                             const float* __restrict__ ea, const float* __restrict__ mea,
                             const float* __restrict__ M,
                             const float* __restrict__ als, const float* __restrict__ ald,
                             float* __restrict__ alpha, uint32_t* __restrict__ mx) {
    int idx = blockIdx.x * blockDim.x + threadIdx.x;
    if (idx >= EF * NH) return;
    int e = idx >> 2, hh = idx & 3;
    int s, d;
    const float* ap;
    if (e < E) { s = src[e]; d = dst[e]; ap = ea + (size_t)e * EDIM; }
    else       { s = d = e - E;          ap = mea + (size_t)(e - E) * EDIM; }
    float ale = 0.f;
    #pragma unroll
    for (int f = 0; f < EDIM; ++f) ale += ap[f] * M[f * NH + hh];
    float a = als[s * NH + hh] + ald[d * NH + hh] + ale;
    a = (a > 0.f) ? a : 0.2f * a;          // leaky_relu, slope 0.2
    alpha[idx] = a;
    atomicMax(&mx[d * NH + hh], enc_f32(a));
}

// exp(alpha - max) + segment-sum(dst); alpha buffer becomes numerator in place
__global__ void k_edge_exp(const int* __restrict__ dst, const uint32_t* __restrict__ mx,
                           float* __restrict__ alpha, float* __restrict__ den) {
    int idx = blockIdx.x * blockDim.x + threadIdx.x;
    if (idx >= EF * NH) return;
    int e = idx >> 2, hh = idx & 3;
    int d = (e < E) ? dst[e] : (e - E);
    float w = __expf(alpha[idx] - dec_f32(mx[d * NH + hh]));
    alpha[idx] = w;
    atomicAdd(&den[d * NH + hh], w);
}

// scatter aggregation: agg[dst,c] += h[src,c] * wnum[e,head]/ (den[dst,head]+eps)
__global__ void __launch_bounds__(256)
k_edge_agg(const int* __restrict__ src, const int* __restrict__ dst,
           const float* __restrict__ h, const float* __restrict__ wnum,
           const float* __restrict__ den, float* __restrict__ agg) {
    int c  = threadIdx.x;
    int hh = c >> 6;
    for (int e = blockIdx.x; e < EF; e += gridDim.x) {
        int s, d;
        if (e < E) { s = src[e]; d = dst[e]; }
        else       { s = d = e - E; }
        float w = wnum[(size_t)e * NH + hh] / (den[(size_t)d * NH + hh] + 1e-16f);
        atomicAdd(&agg[(size_t)d * HC + c], h[(size_t)s * HC + c] * w);
    }
}

// o = relu(agg + b); add accumulate; bf16 cast for next layer input
__global__ void k_finalize(const float* __restrict__ agg, const float* __restrict__ b,
                           float* __restrict__ add, uint16_t* __restrict__ hb, int first) {
    int idx = blockIdx.x * blockDim.x + threadIdx.x;
    if (idx >= N * HC) return;
    int c = idx & (HC - 1);
    float o = fmaxf(agg[idx] + b[c], 0.f);
    add[idx] = first ? o : (add[idx] + o);
    hb[idx]  = f2bf(o);
}

// ---------------------------------------------------------------------------
// SAGPooling score = segsum(add[src] @ w_rel)[dst] + b_rel + add @ w_root
// ---------------------------------------------------------------------------
__global__ void __launch_bounds__(256)
k_score_dots(const float* __restrict__ add, const float* __restrict__ wr,
             const float* __restrict__ wro, float* __restrict__ srel,
             float* __restrict__ sroot) {
    int n    = blockIdx.x * 8 + (threadIdx.x >> 5);
    int lane = threadIdx.x & 31;
    const float* ap = add + (size_t)n * HC + lane * 8;
    float a = 0.f, b = 0.f;
    #pragma unroll
    for (int j = 0; j < 8; ++j) {
        float v = ap[j];
        a += v * wr[lane * 8 + j];
        b += v * wro[lane * 8 + j];
    }
    a = wave_reduce_add(a);
    b = wave_reduce_add(b);
    if (lane == 0) { srel[n] = a; sroot[n] = b; }
}

__global__ void k_score_edges(const int* __restrict__ src, const int* __restrict__ dst,
                              const float* __restrict__ srel, float* __restrict__ aggsc) {
    int e = blockIdx.x * blockDim.x + threadIdx.x;
    if (e >= E) return;
    atomicAdd(&aggsc[dst[e]], srel[src[e]]);
}

__global__ void k_score_final(const float* __restrict__ aggsc, const float* __restrict__ sroot,
                              const float* __restrict__ brel, const int* __restrict__ batch,
                              float* __restrict__ score, uint32_t* __restrict__ cnt) {
    int n = blockIdx.x * blockDim.x + threadIdx.x;
    if (n >= N) return;
    score[n] = aggsc[n] + brel[0] + sroot[n];
    atomicAdd(&cnt[batch[n]], 1u);
}

__global__ void k_scan(const uint32_t* __restrict__ cnt, uint32_t* __restrict__ off) {
    if (threadIdx.x == 0) {
        uint32_t run = 0;
        for (int g = 0; g < NG; ++g) { off[g] = run; run += cnt[g]; }
    }
}

// stable descending rank within each graph (== lexsort((-score, batch)) position)
__global__ void __launch_bounds__(256)
k_rank(const float* __restrict__ score, const uint32_t* __restrict__ cnt,
       const uint32_t* __restrict__ off, int* __restrict__ pos) {
    __shared__ float sl[1024];
    int g = blockIdx.x;
    int start = (int)off[g];
    int n     = (int)cnt[g];
    if (n == 0) return;
    int iters = (n + 255) / 256;
    for (int ii = 0; ii < iters; ++ii) {
        int li    = ii * 256 + threadIdx.x;
        bool valid = li < n;
        int   i  = start + li;
        float si = valid ? score[i] : 0.f;
        int rank = 0;
        for (int j0 = 0; j0 < n; j0 += 1024) {
            __syncthreads();
            int tl = min(1024, n - j0);
            for (int jj = threadIdx.x; jj < tl; jj += 256)
                sl[jj] = score[start + j0 + jj];
            __syncthreads();
            if (valid) {
                for (int jj = 0; jj < tl; ++jj) {
                    float sj = sl[jj];
                    int   j  = start + j0 + jj;
                    rank += (sj > si) || (sj == si && j < i);
                }
            }
        }
        if (valid) pos[i] = start + rank;
    }
}

// out[pos[n]] = sigmoid((add[n]*tanh(score[n])) @ wl + bl)
__global__ void __launch_bounds__(256)
k_out(const float* __restrict__ add, const float* __restrict__ score,
      const int* __restrict__ pos, const float* __restrict__ wl,
      const float* __restrict__ bl, float* __restrict__ out) {
    int n    = blockIdx.x * 8 + (threadIdx.x >> 5);
    int lane = threadIdx.x & 31;
    float t = tanhf(score[n]);
    const float* ap = add + (size_t)n * HC + lane * 8;
    float xp[8];
    #pragma unroll
    for (int j = 0; j < 8; ++j) xp[j] = ap[j] * t;
    int p = pos[n];
    for (int k = 0; k < OUTD; ++k) {
        float s = 0.f;
        #pragma unroll
        for (int j = 0; j < 8; ++j) s += xp[j] * wl[(size_t)(lane * 8 + j) * OUTD + k];
        s = wave_reduce_add(s);
        if (lane == 0) out[(size_t)p * OUTD + k] = 1.f / (1.f + __expf(-(s + bl[k])));
    }
}

// ---------------------------------------------------------------------------
// Host launcher
// ---------------------------------------------------------------------------
static inline int g256(long long n) { return (int)((n + 255) / 256); }

extern "C" void kernel_launch(void* const* d_in, const int* in_sizes, int n_in,
                              void* d_out, int out_size, void* d_ws, size_t ws_size,
                              hipStream_t stream) {
    (void)in_sizes; (void)n_in; (void)out_size; (void)ws_size;

    const float* x     = (const float*)d_in[0];
    const int*   ei    = (const int*)  d_in[1];
    const int*   batch = (const int*)  d_in[2];
    const float* ea    = (const float*)d_in[3];
    const float* W [3] = {(const float*)d_in[4],  (const float*)d_in[10], (const float*)d_in[16]};
    const float* AS[3] = {(const float*)d_in[5],  (const float*)d_in[11], (const float*)d_in[17]};
    const float* AD[3] = {(const float*)d_in[6],  (const float*)d_in[12], (const float*)d_in[18]};
    const float* WE[3] = {(const float*)d_in[7],  (const float*)d_in[13], (const float*)d_in[19]};
    const float* AE[3] = {(const float*)d_in[8],  (const float*)d_in[14], (const float*)d_in[20]};
    const float* B [3] = {(const float*)d_in[9],  (const float*)d_in[15], (const float*)d_in[21]};
    const float* w_rel  = (const float*)d_in[22];
    const float* b_rel  = (const float*)d_in[23];
    const float* w_root = (const float*)d_in[24];
    const float* wl     = (const float*)d_in[25];
    const float* bl     = (const float*)d_in[26];
    float* out = (float*)d_out;

    const int* src = ei;
    const int* dst = ei + E;

    // ---- workspace carve (256B aligned) ----
    char* p = (char*)d_ws;
    auto carve = [&](size_t bytes) -> void* {
        void* r = (void*)p;
        p += (bytes + 255) & ~(size_t)255;
        return r;
    };
    uint16_t* hb    = (uint16_t*)carve((size_t)N * HC * 2);   // bf16 input (x-cast then layer outputs)
    uint16_t* wt    = (uint16_t*)carve((size_t)HC * HC * 2);  // transposed bf16 weights (reused per layer)
    float*    h     = (float*)   carve((size_t)N * HC * 4);   // GEMM output (pre-bias)
    float*    agg   = (float*)   carve((size_t)N * HC * 4);   // attention aggregation target
    float*    add   = (float*)   carve((size_t)N * HC * 4);   // residual accumulator
    float*    als   = (float*)   carve((size_t)N * NH * 4);
    float*    ald   = (float*)   carve((size_t)N * NH * 4);
    float*    wnum  = (float*)   carve((size_t)EF * NH * 4);  // alpha, then exp numerator
    uint32_t* mx    = (uint32_t*)carve((size_t)N * NH * 4);
    float*    den   = (float*)   carve((size_t)N * NH * 4);
    float*    deg   = (float*)   carve((size_t)N * 4);
    float*    mea   = (float*)   carve((size_t)N * EDIM * 4); // sum then mean in place
    float*    Mf    = (float*)   carve((size_t)EDIM * NH * 4);
    float*    srel  = (float*)   carve((size_t)N * 4);
    float*    sroot = (float*)   carve((size_t)N * 4);
    float*    score = (float*)   carve((size_t)N * 4);
    float*    aggsc = (float*)   carve((size_t)N * 4);
    uint32_t* cnt   = (uint32_t*)carve((size_t)NG * 4);
    uint32_t* off   = (uint32_t*)carve((size_t)NG * 4);
    int*      pos   = (int*)     carve((size_t)N * 4);

    // ---- self-loop edge_attr prep ----
    k_zero_u32<<<g256(N), 256, 0, stream>>>((uint32_t*)deg, N);
    k_zero_u32<<<g256((long long)N * EDIM), 256, 0, stream>>>((uint32_t*)mea, N * EDIM);
    k_zero_u32<<<g256(N), 256, 0, stream>>>((uint32_t*)aggsc, N);
    k_zero_u32<<<1, 256, 0, stream>>>(cnt, NG);

    k_deg_sum<<<g256((long long)E * EDIM), 256, 0, stream>>>(dst, ea, deg, mea);
    k_mean_div<<<g256((long long)N * EDIM), 256, 0, stream>>>(mea, deg);

    // x -> bf16
    k_cast_bf16<<<g256((long long)N * FIN), 256, 0, stream>>>(x, hb, N * FIN);

    // ---- 3 GAT layers ----
    for (int l = 0; l < 3; ++l) {
        const int K = (l == 0) ? FIN : HC;

        k_transpose_cast<<<g256((long long)K * HC), 256, 0, stream>>>(W[l], wt, K);
        k_gemm_bf16<<<N / 16, 256, 0, stream>>>(hb, wt, h, K);

        k_fold_M<<<1, 64, 0, stream>>>(WE[l], AE[l], Mf);
        k_node_logits<<<g256((long long)N * NH), 256, 0, stream>>>(h, AS[l], AD[l], als, ald);

        k_zero_u32<<<g256((long long)N * NH), 256, 0, stream>>>(mx, N * NH);
        k_zero_u32<<<g256((long long)N * NH), 256, 0, stream>>>((uint32_t*)den, N * NH);

        k_edge_alpha<<<g256((long long)EF * NH), 256, 0, stream>>>(src, dst, ea, mea, Mf,
                                                                   als, ald, wnum, mx);
        k_edge_exp<<<g256((long long)EF * NH), 256, 0, stream>>>(dst, mx, wnum, den);

        k_zero_u32<<<g256((long long)N * HC), 256, 0, stream>>>((uint32_t*)agg, N * HC);
        k_edge_agg<<<4096, 256, 0, stream>>>(src, dst, h, wnum, den, agg);

        k_finalize<<<g256((long long)N * HC), 256, 0, stream>>>(agg, B[l], add, hb, l == 0);
    }

    // ---- SAGPooling score + final linear ----
    k_score_dots<<<N / 8, 256, 0, stream>>>(add, w_rel, w_root, srel, sroot);
    k_score_edges<<<g256(E), 256, 0, stream>>>(src, dst, srel, aggsc);
    k_score_final<<<g256(N), 256, 0, stream>>>(aggsc, sroot, b_rel, batch, score, cnt);
    k_scan<<<1, 64, 0, stream>>>(cnt, off);
    k_rank<<<NG, 256, 0, stream>>>(score, cnt, off, pos);
    k_out<<<N / 8, 256, 0, stream>>>(add, score, pos, wl, bl, out);
}
